// SSDLoss_24464133718743
// MI455X (gfx1250) — compile-verified, matrix operand order checked by reading, not run on hardware
//
#include <hip/hip_runtime.h>
#include <hip/hip_bf16.h>
#include <math.h>

#define NUM_CLASSES 21
#define BATCH 128
#define ANCHORS 8732
#define APB 128                              // anchors per block in kernel 1
#define N_TOTAL (BATCH * ANCHORS)            // 1,117,696 (divisible by APB)
#define NBLK1 (N_TOTAL / APB)                // 8732 blocks

typedef __attribute__((ext_vector_type(16))) _Float16 v16h;
typedef __attribute__((ext_vector_type(8)))  float    v8f;
typedef __attribute__((ext_vector_type(4)))  unsigned v4u;
typedef __attribute__((ext_vector_type(8)))  int      v8i;
typedef __attribute__((ext_vector_type(4)))  int      v4i;

// ---------------------------------------------------------------- kernel 0
__global__ void k0_init(int* __restrict__ rowcount) {
  const int t = threadIdx.x;
  if (t < BATCH) rowcount[t] = 0;
}

// ---------------------------------------------------------------- kernel 1
// Per-anchor: CE via log-softmax (WMMA row-sum of exp), smooth-L1 loc loss,
// positive counts, per-block partial sums. Logit tile staged by the TDM.
__global__ __launch_bounds__(APB) void k1_per_anchor(
    const float* __restrict__ loc_preds, const float* __restrict__ loc_targets,
    const float* __restrict__ cls_preds, const int* __restrict__ cls_targets,
    float* __restrict__ ce_neg, int* __restrict__ rowcount,
    float* __restrict__ partial_loc, float* __restrict__ partial_posce)
{
  __shared__ float logits[APB * NUM_CLASSES];   // 2688 floats = 10.5 KB
  __shared__ float sumexp[APB];
  __shared__ float maxs[APB];
  __shared__ float redf[APB];

  const int t    = threadIdx.x;
  const int base = blockIdx.x * APB;

  // ---- Stage the 128-anchor x 21-class logit tile into LDS via the TDM ----
#if __has_builtin(__builtin_amdgcn_tensor_load_to_lds) && __has_builtin(__builtin_amdgcn_s_wait_tensorcnt)
  if (t < 32) {
    const unsigned TD0 = APB * NUM_CLASSES;     // 2688 contiguous f32 elements
    const unsigned lds_off =
        (unsigned)(unsigned long long)(uintptr_t)&logits[0];
    const unsigned long long ga =
        (unsigned long long)(uintptr_t)(cls_preds + (size_t)base * NUM_CLASSES);
    // D# group 0: count=1 | lds_addr | global_addr[56:0] | type=2
    v4u g0;
    g0[0] = 1u;
    g0[1] = lds_off;
    g0[2] = (unsigned)(ga & 0xFFFFFFFFull);
    g0[3] = (unsigned)((ga >> 32) & 0x1FFFFFFull) | 0x80000000u;  // type=2
    // D# group 1: data_size=4B, 2D tensor TD0 x 1, tile TD0 x 1
    v8i g1;
    g1[0] = (int)(2u << 16);          // data_size = 2 (4 bytes), mask=0
    g1[1] = (int)(TD0 << 16);         // tensor_dim0[15:0]
    g1[2] = (int)(1u << 16);          // tensor_dim1 = 1
    g1[3] = (int)(TD0 << 16);         // tile_dim0
    g1[4] = 1;                        // tile_dim1 = 1, tile_dim2 = 0
    g1[5] = (int)TD0;                 // tensor_dim0_stride[31:0]
    g1[6] = (int)(TD0 << 16);         // tensor_dim1_stride[15:0]
    g1[7] = 0;
    v4i z4; z4[0] = z4[1] = z4[2] = z4[3] = 0;
    v8i z8;
    #pragma unroll
    for (int e = 0; e < 8; ++e) z8[e] = 0;
    __builtin_amdgcn_tensor_load_to_lds(g0, g1, z4, z4, z8, 0);
    __builtin_amdgcn_s_wait_tensorcnt(0);
  }
#else
  for (int i = t; i < APB * NUM_CLASSES; i += APB)
    logits[i] = cls_preds[(size_t)base * NUM_CLASSES + i];
#endif
  __syncthreads();

  // ---- sum of exp per anchor via v_wmma_f32_16x16x32_f16 (B = ones) ----
  {
    const int lane = t & 31;
    const int wv   = t >> 5;
    for (int g = 0; g < 2; ++g) {
      const int m0 = wv * 32 + g * 16;         // 16 anchors per WMMA
      const int m  = m0 + (lane & 15);
      const float* lp = &logits[m * NUM_CLASSES];
      float mx = lp[0];
      #pragma unroll
      for (int c = 1; c < NUM_CLASSES; ++c) mx = fmaxf(mx, lp[c]);

      // ISA 16-bit A-matrix 16x32 layout: lane<16 holds K {0..7,16..23},
      // lane>=16 holds K {8..15,24..31}; element e=2j,2j+1 -> VGPR j halves.
      const int kbase = (lane < 16) ? 0 : 8;
      v16h a;
      #pragma unroll
      for (int e = 0; e < 8; ++e) {
        const int k = kbase + e;                       // always < 16 < 21
        a[e] = (_Float16)__expf(lp[k] - mx);
      }
      #pragma unroll
      for (int e = 8; e < 16; ++e) {
        const int k = 16 + kbase + (e - 8);            // 16..31
        a[e] = (k < NUM_CLASSES) ? (_Float16)__expf(lp[k] - mx)
                                 : (_Float16)0.f;
      }
      v16h bones;
      #pragma unroll
      for (int e = 0; e < 16; ++e) bones[e] = (_Float16)1.0f;
      v8f c = {};
      const v8f d = __builtin_amdgcn_wmma_f32_16x16x32_f16(
          false, a, false, bones, (short)0, c, false, false);

      if (lane < 16) maxs[m0 + lane] = mx;
      // D layout: VGPR r holds M=r (lanes 0-15) / M=r+8 (lanes 16-31);
      // every column equals the row-sum, so lanes 0 and 16 scatter it.
      if ((lane & 15) == 0) {
        const int mb = m0 + ((lane >> 4) << 3);
        #pragma unroll
        for (int r = 0; r < 8; ++r) sumexp[mb + r] = d[r];
      }
    }
  }
  __syncthreads();

  // ---- finish per-anchor: CE, pos mask, smooth-L1 ----
  const size_t idx  = (size_t)base + t;
  const int    traw = cls_targets[idx];
  const int    tgt  = min(max(traw, 0), NUM_CLASSES - 1);
  float ce = __logf(sumexp[t]) + maxs[t] - logits[t * NUM_CLASSES + tgt];
  if (traw < 0) ce = 0.f;
  const bool pos = traw > 0;
  ce_neg[idx] = pos ? -1.0f : ce;              // sentinel -1 marks positives

  const float4 p = ((const float4*)loc_preds)[idx];
  const float4 q = ((const float4*)loc_targets)[idx];
  float l = 0.f, dx;
  dx = fabsf(p.x - q.x); l += (dx < 1.f) ? 0.5f * dx * dx : dx - 0.5f;
  dx = fabsf(p.y - q.y); l += (dx < 1.f) ? 0.5f * dx * dx : dx - 0.5f;
  dx = fabsf(p.z - q.z); l += (dx < 1.f) ? 0.5f * dx * dx : dx - 0.5f;
  dx = fabsf(p.w - q.w); l += (dx < 1.f) ? 0.5f * dx * dx : dx - 0.5f;
  if (!pos) l = 0.f;

  const int b = (int)(idx / ANCHORS);
  if (pos) atomicAdd(&rowcount[b], 1);         // integer atomic: deterministic

  // fixed-order block reductions -> per-block partials (no float atomics)
  redf[t] = l; __syncthreads();
  for (int off = APB / 2; off > 0; off >>= 1) {
    if (t < off) redf[t] += redf[t + off];
    __syncthreads();
  }
  if (t == 0) partial_loc[blockIdx.x] = redf[0];
  __syncthreads();
  redf[t] = pos ? ce : 0.f; __syncthreads();
  for (int off = APB / 2; off > 0; off >>= 1) {
    if (t < off) redf[t] += redf[t + off];
    __syncthreads();
  }
  if (t == 0) partial_posce[blockIdx.x] = redf[0];
}

// ---------------------------------------------------------------- kernel 2
// Per-row hard-negative mining: sum of the top-(3*num_pos) negative CE values
// via 4-pass LDS radix-select (replaces the reference's double argsort).
__global__ __launch_bounds__(256) void k2_rowselect(
    const float* __restrict__ ce_neg, const int* __restrict__ rowcount,
    float* __restrict__ row_negsum)
{
  __shared__ float    vals[ANCHORS];          // 34.9 KB of the 320 KB WGP LDS
  __shared__ unsigned hist[256];
  __shared__ float    red[256];
  __shared__ unsigned sh_digit;
  __shared__ int      sh_rem;

  const int t = threadIdx.x;
  const int b = blockIdx.x;
  const float* row = ce_neg + (size_t)b * ANCHORS;
  for (int i = t; i < ANCHORS; i += 256) vals[i] = row[i];

  const int npos   = rowcount[b];
  const int numneg = ANCHORS - npos;
  const long long want = 3LL * (long long)npos;
  const int take = (want < (long long)numneg) ? (int)want : numneg;
  __syncthreads();
  if (take <= 0) { if (t == 0) row_negsum[b] = 0.f; return; }

  // keys: 0 for sentinels/zeros, raw IEEE bits for ce>0 (monotonic)
  unsigned prefix = 0, prefmask = 0;
  int remaining = take;
  for (int shift = 24; shift >= 0; shift -= 8) {
    hist[t] = 0;
    __syncthreads();
    for (int i = t; i < ANCHORS; i += 256) {
      const float v = vals[i];
      const unsigned key = (v <= 0.f) ? 0u : __float_as_uint(v);
      if ((key & prefmask) == prefix)
        atomicAdd(&hist[(key >> shift) & 255u], 1u);
    }
    __syncthreads();
    if (t == 0) {
      unsigned cum = 0; int d = 255;
      for (; d > 0; --d) {
        if (cum + hist[d] >= (unsigned)remaining) break;
        cum += hist[d];
      }
      sh_digit = (unsigned)d;
      sh_rem   = remaining - (int)cum;
    }
    __syncthreads();
    prefix   |= sh_digit << shift;
    prefmask |= 0xFFu << shift;
    remaining = sh_rem;
    __syncthreads();
  }

  const unsigned T = prefix;                  // take-th largest key
  float s = 0.f;
  for (int i = t; i < ANCHORS; i += 256) {
    const float v = vals[i];
    const unsigned key = (v <= 0.f) ? 0u : __float_as_uint(v);
    if (key > T) s += v;
  }
  red[t] = s; __syncthreads();
  for (int off = 128; off > 0; off >>= 1) {
    if (t < off) red[t] += red[t + off];
    __syncthreads();
  }
  if (t == 0) {
    const float tval = (T == 0u) ? 0.f : __uint_as_float(T);
    row_negsum[b] = red[0] + tval * (float)remaining;   // ties at threshold
  }
}

// ---------------------------------------------------------------- kernel 3
__global__ __launch_bounds__(256) void k3_final(
    const float* __restrict__ partial_loc, const float* __restrict__ partial_posce,
    const float* __restrict__ row_negsum, const int* __restrict__ rowcount,
    float* __restrict__ out)
{
  __shared__ float red[256];
  __shared__ int   redi[256];
  const int t = threadIdx.x;
  float s = 0.f; int np = 0;
  for (int i = t; i < NBLK1; i += 256) s += partial_loc[i] + partial_posce[i];
  for (int i = t; i < BATCH; i += 256) { s += row_negsum[i]; np += rowcount[i]; }
  red[t] = s; redi[t] = np; __syncthreads();
  for (int off = 128; off > 0; off >>= 1) {
    if (t < off) { red[t] += red[t + off]; redi[t] += redi[t + off]; }
    __syncthreads();
  }
  if (t == 0) out[0] = red[0] / (float)redi[0];
}

// ---------------------------------------------------------------- launcher
extern "C" void kernel_launch(void* const* d_in, const int* in_sizes, int n_in,
                              void* d_out, int out_size, void* d_ws, size_t ws_size,
                              hipStream_t stream) {
  (void)in_sizes; (void)n_in; (void)out_size; (void)ws_size;
  const float* loc_preds   = (const float*)d_in[0];
  const float* loc_targets = (const float*)d_in[1];
  const float* cls_preds   = (const float*)d_in[2];
  const int*   cls_targets = (const int*)d_in[3];
  float* out = (float*)d_out;

  // workspace layout (~4.55 MB)
  float* ce_neg        = (float*)d_ws;                       // N_TOTAL f32
  int*   rowcount      = (int*)(ce_neg + N_TOTAL);           // BATCH i32
  float* partial_loc   = (float*)(rowcount + BATCH);         // NBLK1 f32
  float* partial_posce = partial_loc + NBLK1;                // NBLK1 f32
  float* row_negsum    = partial_posce + NBLK1;              // BATCH f32

  k0_init<<<1, BATCH, 0, stream>>>(rowcount);
  k1_per_anchor<<<NBLK1, APB, 0, stream>>>(loc_preds, loc_targets, cls_preds,
                                           cls_targets, ce_neg, rowcount,
                                           partial_loc, partial_posce);
  k2_rowselect<<<BATCH, 256, 0, stream>>>(ce_neg, rowcount, row_negsum);
  k3_final<<<1, 256, 0, stream>>>(partial_loc, partial_posce, row_negsum,
                                  rowcount, out);
}
